// RVAE_24601572671735
// MI455X (gfx1250) — matrix operand order repaired and checked by reading, not compiled, not run on hardware
//
// RVAE on MI455X (gfx1250): persistent weight-stationary WMMA matvec chain.
//
//  - 16 persistent workgroups x 256 threads (8 wave32s), one per WGP.
//  - Recurrence weights fp32->bf16 once into LDS (256KB slice/WGP).
//  - Per step: stage x (bf16) into LDS, v_wmma_f32_16x16x32_bf16 over the
//    WG's row slice with 4 independent accumulators (breaks the WMMA RAW
//    chain), results staged through a 512B LDS buffer so tanh+bias+store is
//    one row per thread, then one device-scope release/acquire barrier.
//  - Row-partitioned across WGs => no cross-WG reduction, 1 barrier/step.
#include <hip/hip_runtime.h>
#include <hip/hip_bf16.h>
#include <cstddef>

#define WORDN 1024
#define HIDN  2048
#define ENCN  512
#define NODEN 8192
#define NWG   16
#define TPB   256
#define KENC  (2 * WORDN)              // 2048
#define ENC_ROWS (WORDN / NWG)         // 64 rows/WG (encode)
#define DEC_ROWS ((2 * WORDN) / NWG)   // 128 rows/WG (decode)
#define WSLICE (ENC_ROWS * KENC)       // 131072 bf16 = 256KB (== DEC_ROWS*WORDN)

typedef __bf16    v16bf __attribute__((ext_vector_type(16)));
typedef float     v8f   __attribute__((ext_vector_type(8)));
typedef unsigned  u32x4 __attribute__((ext_vector_type(4)));

union FragU { v16bf v; u32x4 q[2]; };

// ---- workspace layout in d_ws --------------------------------------------
struct Ctrl {
  unsigned cnt, gen; unsigned pad[14];   // global barrier state (zeroed at launch)
  float P0[WORDN], P1[WORDN];            // encode parent ping-pong
  float hid[HIDN];
  float mu[ENCN], va[ENCN], z[ENCN];
  float h2[HIDN];
  float D0[WORDN], D1[WORDN];            // decode parent ping-pong
};

__device__ __forceinline__ unsigned short f2bf(float f) {
  unsigned u = __float_as_uint(f);                   // RNE round to bf16
  return (unsigned short)((u + 0x7FFFu + ((u >> 16) & 1u)) >> 16);
}

// A fragment: 16x32 bf16 tile of row-major LDS weights, per ISA 16-bit A layout.
// lanes 0-15: K {0..7,16..23} of row M=lane; lanes 16-31: K {8..15,24..31}.
__device__ __forceinline__ v16bf load_a(const unsigned short* wl, int K,
                                        int rowbase, int kb, int lane) {
  const unsigned short* p =
      wl + (size_t)(rowbase + (lane & 15)) * K + kb + ((lane >> 4) << 3);
  FragU f;
  f.q[0] = *(const u32x4*)(p);        // ds_read_b128
  f.q[1] = *(const u32x4*)(p + 16);   // ds_read_b128
  return f.v;
}

// B fragment: x replicated into all 16 columns (same K order as A:
// lanes 0-15 hold K {0..7,16..23} for col N=lane, lanes 16-31 hold K {8..15,24..31}).
__device__ __forceinline__ v16bf load_b(const unsigned short* xl, int kb, int lane) {
  const unsigned short* p = xl + kb + ((lane >> 4) << 3);
  FragU f;
  f.q[0] = *(const u32x4*)(p);
  f.q[1] = *(const u32x4*)(p + 16);
  return f.v;
}

__device__ __forceinline__ v8f wmma_tile(const unsigned short* wl, int K, int rbase,
                                         const unsigned short* xl, int kb, int lane,
                                         v8f acc) {
  v16bf a = load_a(wl, K, rbase, kb, lane);
  v16bf b = load_b(xl, kb, lane);
  return __builtin_amdgcn_wmma_f32_16x16x32_bf16(false, a, false, b,
                                                 (short)0, acc, false, false);
}

// 16x(klen) x-slice dot for 16 rows starting at rbase; 4 independent
// accumulators so the v_wmma chain is issue/LDS-bound, not latency-bound.
__device__ __forceinline__ v8f matvec_rows16(const unsigned short* wl, int K,
                                             int rbase, const unsigned short* xl,
                                             int k0, int klen, int lane) {
  v8f a0 = {0.f,0.f,0.f,0.f,0.f,0.f,0.f,0.f};
  v8f a1 = a0, a2 = a0, a3 = a0;
  for (int kb = k0; kb < k0 + klen; kb += 128) {
    a0 = wmma_tile(wl, K, rbase, xl, kb,      lane, a0);
    a1 = wmma_tile(wl, K, rbase, xl, kb + 32, lane, a1);
    a2 = wmma_tile(wl, K, rbase, xl, kb + 64, lane, a2);
    a3 = wmma_tile(wl, K, rbase, xl, kb + 96, lane, a3);
  }
  return (a0 + a1) + (a2 + a3);
}

// Device-wide sense-reversing barrier (all 16 WGs are co-resident).
__device__ __forceinline__ void gbar(Ctrl* ws) {
  __syncthreads();
  if (threadIdx.x == 0) {
    unsigned g = __hip_atomic_load(&ws->gen, __ATOMIC_RELAXED, __HIP_MEMORY_SCOPE_AGENT);
    unsigned a = __hip_atomic_fetch_add(&ws->cnt, 1u, __ATOMIC_ACQ_REL, __HIP_MEMORY_SCOPE_AGENT);
    if (a == NWG - 1) {
      __hip_atomic_store(&ws->cnt, 0u, __ATOMIC_RELAXED, __HIP_MEMORY_SCOPE_AGENT);
      __hip_atomic_store(&ws->gen, g + 1u, __ATOMIC_RELEASE, __HIP_MEMORY_SCOPE_AGENT);
    } else {
      while (__hip_atomic_load(&ws->gen, __ATOMIC_ACQUIRE, __HIP_MEMORY_SCOPE_AGENT) == g)
        __builtin_amdgcn_s_sleep(1);
    }
  }
  __syncthreads();
}

// One-shot small matvec, fp32 weights streamed from L2, grid-strided rows.
__device__ __forceinline__ void matvec_global(const float* __restrict__ W,
                                              const float* __restrict__ b,
                                              const float* __restrict__ x,
                                              float* __restrict__ y,
                                              int M, int K, int gtid, int gsz) {
  for (int m = gtid; m < M; m += gsz) {
    float s = b[m];
    const float* wr = W + (size_t)m * K;
    for (int k = 0; k < K; ++k) s = fmaf(wr[k], x[k], s);
    y[m] = tanhf(s);
  }
}

__global__ void __launch_bounds__(TPB, 1) rvae_persistent(
    const float* __restrict__ sent,    const float* __restrict__ eps,
    const float* __restrict__ WeC2P_w, const float* __restrict__ WeC2P_b,
    const float* __restrict__ WeP2H_w, const float* __restrict__ WeP2H_b,
    const float* __restrict__ WeH2M_w, const float* __restrict__ WeH2M_b,
    const float* __restrict__ WeH2D_w, const float* __restrict__ WeH2D_b,
    const float* __restrict__ WdE2H_w, const float* __restrict__ WdE2H_b,
    const float* __restrict__ WdH2P_w, const float* __restrict__ WdH2P_b,
    const float* __restrict__ WdP2C_w, const float* __restrict__ WdP2C_b,
    float* __restrict__ out, Ctrl* ws) {
  extern __shared__ char smem[];
  unsigned short* wl  = (unsigned short*)smem;                       // 256KB weights
  unsigned short* xl  = (unsigned short*)(smem + WSLICE * 2);        // 4KB x (bf16)
  float*          red = (float*)(smem + WSLICE * 2 + KENC * 2);      // 512B results

  const int wg   = blockIdx.x;
  const int wv   = threadIdx.x >> 5;
  const int lane = threadIdx.x & 31;
  const int gtid = wg * TPB + threadIdx.x;
  const int gsz  = NWG * TPB;

  // ---- preload encode weight slice (rows wg*64 .. wg*64+63 of WeC2P) -----
  {
    const float* src = WeC2P_w + (size_t)wg * ENC_ROWS * KENC;
    for (int i = threadIdx.x; i < WSLICE; i += TPB) wl[i] = f2bf(src[i]);
  }
  // initial parent = sent[0]
  for (int i = gtid; i < WORDN; i += gsz) ws->P1[i] = sent[i];
  gbar(ws);

  // ======================= ENCODE: 8191 sequential matvecs ================
  const float* pin  = ws->P1;
  float*       pout = ws->P0;
  for (int t = 1; t < NODEN; ++t) {
    const float* srow = sent + (size_t)t * WORDN;
    if (t + 1 < NODEN)
      __builtin_prefetch(sent + (size_t)(t + 1) * WORDN + (threadIdx.x << 2), 0, 1);
    for (int i = threadIdx.x; i < KENC; i += TPB) {
      float v = (i < WORDN) ? pin[i] : srow[i - WORDN];
      xl[i] = f2bf(v);
    }
    __syncthreads();

    // 4 row-groups x 2 K-halves across the 8 waves
    const int rbase = (wv & 3) << 4;
    const int k0    = (wv >> 2) * (KENC / 2);
    v8f acc = matvec_rows16(wl, KENC, rbase, xl, k0, KENC / 2, lane);

    // column-replicated D: lanes 0/16 own rows rbase+0..7 / rbase+8..15
    if (lane == 0 || lane == 16) {
      float* rw = red + wv * 16 + ((lane >> 4) << 3);
#pragma unroll
      for (int j = 0; j < 8; ++j) rw[j] = acc[j];
    }
    __syncthreads();
    // one row per thread: sum the two K-half partials, bias+tanh, store
    if (threadIdx.x < ENC_ROWS) {
      const int g  = threadIdx.x >> 4;       // row-group
      const int jl = threadIdx.x & 15;       // row within group
      const int r  = wg * ENC_ROWS + threadIdx.x;
      float s = red[g * 16 + jl] + red[(g + 4) * 16 + jl] + WeC2P_b[r];
      pout[r] = tanhf(s);
    }
    gbar(ws);
    const float* tp = pin; pin = pout; pout = (float*)tp;
  }

  // ======================= middle one-shot section ========================
  matvec_global(WeP2H_w, WeP2H_b, pin, ws->hid, HIDN, WORDN, gtid, gsz);
  gbar(ws);
  matvec_global(WeH2M_w, WeH2M_b, ws->hid, ws->mu, ENCN, HIDN, gtid, gsz);
  matvec_global(WeH2D_w, WeH2D_b, ws->hid, ws->va, ENCN, HIDN, gtid, gsz);
  gbar(ws);
  for (int e = gtid; e < ENCN; e += gsz)
    ws->z[e] = eps[e] * expf(0.5f * ws->va[e]) + ws->mu[e];
  gbar(ws);
  matvec_global(WdE2H_w, WdE2H_b, ws->z, ws->h2, HIDN, ENCN, gtid, gsz);
  gbar(ws);
  matvec_global(WdH2P_w, WdH2P_b, ws->h2, ws->D1, WORDN, HIDN, gtid, gsz);
  gbar(ws);

  // ---- swap LDS weights to decode slice (rows wg*128.. of WdP2C) ---------
  {
    const float* src = WdP2C_w + (size_t)wg * DEC_ROWS * WORDN;
    for (int i = threadIdx.x; i < WSLICE; i += TPB) wl[i] = f2bf(src[i]);
  }
  __syncthreads();

  // ======================= DECODE: 8191 sequential matvecs ================
  pin = ws->D1; pout = ws->D0;
  for (int i = 0; i < NODEN - 1; ++i) {
    for (int k = threadIdx.x; k < WORDN; k += TPB) xl[k] = f2bf(pin[k]);
    __syncthreads();

    // 8 waves x 16 rows, full K each (no cross-wave reduce)
    v8f acc = matvec_rows16(wl, WORDN, wv << 4, xl, 0, WORDN, lane);

    if (lane == 0 || lane == 16) {
      float* rw = red + wv * 16 + ((lane >> 4) << 3);
#pragma unroll
      for (int j = 0; j < 8; ++j) rw[j] = acc[j];
    }
    __syncthreads();
    if (threadIdx.x < DEC_ROWS) {           // one row per thread
      const int r = wg * DEC_ROWS + threadIdx.x;
      float y = tanhf(red[threadIdx.x] + WdP2C_b[r]);
      if (r < WORDN) out[(size_t)(NODEN - 1 - i) * WORDN + r] = y;   // ys[i]
      else           pout[r - WORDN] = y;                            // next p
    }
    gbar(ws);
    const float* tp = pin; pin = pout; pout = (float*)tp;
  }
  // out row 0 = final carry
  for (int k = gtid; k < WORDN; k += gsz) out[k] = pin[k];
}

extern "C" void kernel_launch(void* const* d_in, const int* in_sizes, int n_in,
                              void* d_out, int out_size, void* d_ws, size_t ws_size,
                              hipStream_t stream) {
  (void)in_sizes; (void)n_in; (void)out_size; (void)ws_size;
  const float* sent    = (const float*)d_in[0];
  const float* eps     = (const float*)d_in[1];
  const float* WeC2P_w = (const float*)d_in[2];
  const float* WeC2P_b = (const float*)d_in[3];
  const float* WeP2H_w = (const float*)d_in[4];
  const float* WeP2H_b = (const float*)d_in[5];
  const float* WeH2M_w = (const float*)d_in[6];
  const float* WeH2M_b = (const float*)d_in[7];
  const float* WeH2D_w = (const float*)d_in[8];
  const float* WeH2D_b = (const float*)d_in[9];
  const float* WdE2H_w = (const float*)d_in[10];
  const float* WdE2H_b = (const float*)d_in[11];
  const float* WdH2P_w = (const float*)d_in[12];
  const float* WdH2P_b = (const float*)d_in[13];
  const float* WdP2C_w = (const float*)d_in[14];
  const float* WdP2C_b = (const float*)d_in[15];
  Ctrl* ws = (Ctrl*)d_ws;

  const size_t smem = (size_t)WSLICE * 2 + (size_t)KENC * 2 + 128 * 4; // 266752B
  hipFuncSetAttribute(reinterpret_cast<const void*>(rvae_persistent),
                      hipFuncAttributeMaxDynamicSharedMemorySize, (int)smem);
  hipMemsetAsync(d_ws, 0, 64, stream);  // reset global-barrier state (deterministic)
  rvae_persistent<<<NWG, TPB, smem, stream>>>(
      sent, eps, WeC2P_w, WeC2P_b, WeP2H_w, WeP2H_b, WeH2M_w, WeH2M_b,
      WeH2D_w, WeH2D_b, WdE2H_w, WdE2H_b, WdH2P_w, WdH2P_b, WdP2C_w, WdP2C_b,
      (float*)d_out, ws);
}
